// HierarchicalAttention_10892037063094
// MI455X (gfx1250) — compile-verified, matrix operand order checked by reading, not run on hardware
//
#include <hip/hip_runtime.h>
#include <hip/hip_bf16.h>

typedef __bf16 bf16_t;
typedef __attribute__((ext_vector_type(16))) __bf16 v16bf;
typedef __attribute__((ext_vector_type(8)))  __bf16 v8bf;
typedef __attribute__((ext_vector_type(8)))  float  v8f;

#define D_FEAT 512
#define R_DIM  128
#define L_LVL  3
#define N_CUR  4096
#define N_HIST 8192
#define N_ALL  (N_CUR + N_HIST)   // 12288: current + historical rows fused

// ---------------------------------------------------------------------------
// Operand loader for NT GEMM (C = A * B^T, both row-major, K contiguous).
// 16x32 bf16 tile per the CDNA5 16-bit A/B layout:
//   lane l (row = l&15) holds K = koff..koff+7 and K = 16+koff..16+koff+7,
//   koff = (l>=16) ? 8 : 0.  Two 16-byte loads, shuffled into one v16bf.
// ---------------------------------------------------------------------------
__device__ __forceinline__ v16bf load_nt(const bf16_t* base, int ld, int lane) {
    const bf16_t* p = base + (size_t)(lane & 15) * (size_t)ld + (size_t)(((lane >> 4) & 1) * 8);
    v8bf lo = *(const v8bf*)p;
    v8bf hi = *(const v8bf*)(p + 16);
    return __builtin_shufflevector(lo, hi, 0,1,2,3,4,5,6,7,8,9,10,11,12,13,14,15);
}

__device__ __forceinline__ v8f wmma_bf16(v16bf a, v16bf b, v8f c) {
    return __builtin_amdgcn_wmma_f32_16x16x32_bf16(false, a, false, b, (short)0, c, false, false);
}

// ---------------------------------------------------------------------------
// fp32 -> bf16 elementwise convert
// ---------------------------------------------------------------------------
__global__ void cvt_f32_bf16(const float* __restrict__ in, bf16_t* __restrict__ out, long n) {
    long i = (long)blockIdx.x * blockDim.x + threadIdx.x;
    long stride = (long)gridDim.x * blockDim.x;
    for (; i < n; i += stride) out[i] = (bf16_t)in[i];
}

// hist [8192,512] f32 -> histT [512,8192] bf16 (coalesced writes; reads hit L2)
__global__ void transpose_hist(const float* __restrict__ hist, bf16_t* __restrict__ histT) {
    long total = (long)D_FEAT * N_HIST;
    long i = (long)blockIdx.x * blockDim.x + threadIdx.x;
    long stride = (long)gridDim.x * blockDim.x;
    for (; i < total; i += stride) {
        int m = (int)(i & (N_HIST - 1));
        int d = (int)(i >> 13);
        histT[i] = (bf16_t)hist[(size_t)m * D_FEAT + d];
    }
}

// ---------------------------------------------------------------------------
// Generic NT GEMM: C[M,N](bf16) = act(A[M,K] * B[N,K]^T + bias), per-level z.
// Block = 128 threads = 4 waves; each wave computes a 32(M) x 64(N) tile.
// K loop unrolled x2 with ping-pong operand sets: one set is consumed by
// WMMAs while the other set's global loads are in flight — no register
// copies between stages.
// ---------------------------------------------------------------------------
__global__ __launch_bounds__(128) void gemm_nt(
    const bf16_t* __restrict__ A, const bf16_t* __restrict__ B,
    const float* __restrict__ bias, bf16_t* __restrict__ C,
    int N, int K, int lda, int ldb,
    long sA, long sB, long sBias, long sC, int relu)
{
    int lane = threadIdx.x & 31;
    int wave = threadIdx.x >> 5;
    int l = blockIdx.z;
    A += (size_t)l * sA;  B += (size_t)l * sB;
    bias += (size_t)l * sBias;  C += (size_t)l * sC;

    int m0 = (blockIdx.y * 4 + wave) * 32;
    int n0 = blockIdx.x * 64;

    v8f acc[2][4];
    #pragma unroll
    for (int u = 0; u < 2; ++u)
        #pragma unroll
        for (int t = 0; t < 4; ++t)
            #pragma unroll
            for (int j = 0; j < 8; ++j) acc[u][t][j] = 0.f;

    const bf16_t* Ar0 = A + (size_t)m0 * lda;
    const bf16_t* Ar1 = A + (size_t)(m0 + 16) * lda;
    const bf16_t* Br[4];
    #pragma unroll
    for (int t = 0; t < 4; ++t) Br[t] = B + (size_t)(n0 + t * 16) * ldb;

    auto loadS = [&](v16bf& a0, v16bf& a1, v16bf* b, int k) {
        a0 = load_nt(Ar0 + k, lda, lane);
        a1 = load_nt(Ar1 + k, lda, lane);
        #pragma unroll
        for (int t = 0; t < 4; ++t) b[t] = load_nt(Br[t] + k, ldb, lane);
    };
    auto mmaS = [&](v16bf& a0, v16bf& a1, v16bf* b) {
        #pragma unroll
        for (int t = 0; t < 4; ++t) {
            acc[0][t] = wmma_bf16(a0, b[t], acc[0][t]);
            acc[1][t] = wmma_bf16(a1, b[t], acc[1][t]);
        }
    };

    // K is a multiple of 64 for all uses (512, 128).
    v16bf a0x, a1x, bx[4], a0y, a1y, by[4];
    loadS(a0x, a1x, bx, 0);
    int k = 0;
    for (; k < K - 64; k += 64) {
        loadS(a0y, a1y, by, k + 32);
        mmaS(a0x, a1x, bx);
        loadS(a0x, a1x, bx, k + 64);
        mmaS(a0y, a1y, by);
    }
    loadS(a0y, a1y, by, k + 32);
    mmaS(a0x, a1x, bx);
    mmaS(a0y, a1y, by);

    int col0 = lane & 15, half = (lane >> 4) & 1;
    #pragma unroll
    for (int u = 0; u < 2; ++u)
        #pragma unroll
        for (int t = 0; t < 4; ++t) {
            int col = n0 + t * 16 + col0;
            float bv = bias[col];
            #pragma unroll
            for (int v = 0; v < 8; ++v) {
                int r = m0 + u * 16 + v + half * 8;
                float val = acc[u][t][v] + bv;
                if (relu) val = fmaxf(val, 0.f);
                C[(size_t)r * N + col] = (bf16_t)val;
            }
        }
}

// ---------------------------------------------------------------------------
// Flash attention per (level, 16-row query tile). Block = 4 waves.
// Wave w: computes 16x16 score sub-tile for columns [j*64+w*16, ..+16) and
// owns the 128-wide output D-slice [w*128, w*128+128) (8 f32 acc tiles).
// Online softmax state in LDS; P tile transits LDS (C-layout -> A-layout).
// Score and PV loops use 2-deep ping-pong operand buffers; next chunk's
// K/V tiles are pulled toward the WGP with global_prefetch.
// ---------------------------------------------------------------------------
__global__ __launch_bounds__(128) void flash_attn(
    const bf16_t* __restrict__ proj,    // [L, N_ALL, D]: rows 0..4095 = cur, 4096.. = hist
    const bf16_t* __restrict__ histT,   // [D, N_HIST]
    bf16_t* __restrict__ lvlout)        // [L, N_CUR, D]
{
    int lane = threadIdx.x & 31;
    int wave = threadIdx.x >> 5;
    int half = (lane >> 4) & 1;
    int col0 = lane & 15;
    int l  = blockIdx.z;
    int m0 = blockIdx.x * 16;

    const bf16_t* pc = proj + ((size_t)l * N_ALL + m0) * D_FEAT;
    const bf16_t* ph = proj + ((size_t)l * N_ALL + N_CUR) * D_FEAT;

    __shared__ float m_i[16], l_i[16], corr[16], red[4][16];
    __shared__ __align__(16) bf16_t Pbuf[16][80];   // 64 cols + pad (160B row stride)

    if (threadIdx.x < 16) { m_i[threadIdx.x] = -3.0e38f; l_i[threadIdx.x] = 0.f; }
    __syncthreads();

    // Register-cache the 16x512 query tile (16 K-steps of v16bf = 128 VGPRs).
    v16bf areg[16];
    #pragma unroll
    for (int k = 0; k < 16; ++k) areg[k] = load_nt(pc + k * 32, D_FEAT, lane);

    v8f acc[8];
    #pragma unroll
    for (int t = 0; t < 8; ++t)
        #pragma unroll
        for (int j = 0; j < 8; ++j) acc[t][j] = 0.f;

    const float scale = 0.04419417382415922f;   // 1/sqrt(512)

    for (int j = 0; j < N_HIST / 64; ++j) {
        int n0 = j * 64 + wave * 16;
        const bf16_t* B = ph + (size_t)n0 * D_FEAT;

        // prefetch next iteration's K rows (16 rows x 1KB = 128 cachelines)
        // and V slab (128 histT rows, one 128B line each) toward the WGP
        if (j + 1 < N_HIST / 64) {
            const char* nb = (const char*)(ph + (size_t)(n0 + 64) * D_FEAT);
            #pragma unroll
            for (int p = 0; p < 4; ++p)
                __builtin_prefetch(nb + (size_t)(p * 32 + lane) * 128, 0, 1);
            const char* nv = (const char*)(histT + (size_t)(wave * 128) * N_HIST + (j + 1) * 64);
            #pragma unroll
            for (int p = 0; p < 4; ++p)
                __builtin_prefetch(nv + (size_t)(p * 32 + lane) * (N_HIST * 2), 0, 1);
        }

        // -------- scores: 16x16 tile, K = 512, ping-pong pipelined --------
        v8f s;
        #pragma unroll
        for (int v = 0; v < 8; ++v) s[v] = 0.f;
        {
            v16bf bq0 = load_nt(B, D_FEAT, lane);
            v16bf bq1;
            #pragma unroll
            for (int k = 0; k < 14; k += 2) {
                bq1 = load_nt(B + (k + 1) * 32, D_FEAT, lane);
                s = wmma_bf16(areg[k], bq0, s);
                bq0 = load_nt(B + (k + 2) * 32, D_FEAT, lane);
                s = wmma_bf16(areg[k + 1], bq1, s);
            }
            bq1 = load_nt(B + 15 * 32, D_FEAT, lane);
            s = wmma_bf16(areg[14], bq0, s);
            s = wmma_bf16(areg[15], bq1, s);
        }
        #pragma unroll
        for (int v = 0; v < 8; ++v) s[v] *= scale;

        // -------- per-row max within this wave's tile --------
        #pragma unroll
        for (int v = 0; v < 8; ++v) {
            float r = s[v];
            r = fmaxf(r, __shfl_xor(r, 1, 32));
            r = fmaxf(r, __shfl_xor(r, 2, 32));
            r = fmaxf(r, __shfl_xor(r, 4, 32));
            r = fmaxf(r, __shfl_xor(r, 8, 32));
            if (col0 == 0) red[wave][v + half * 8] = r;
        }
        __syncthreads();

        if (threadIdx.x < 16) {
            int r = threadIdx.x;
            float nm = fmaxf(fmaxf(fmaxf(red[0][r], red[1][r]),
                                   fmaxf(red[2][r], red[3][r])), m_i[r]);
            corr[r] = __expf(m_i[r] - nm);
            m_i[r]  = nm;
        }
        __syncthreads();

        // -------- P = exp(s - m), row sums, stash P to LDS in row-major --------
        #pragma unroll
        for (int v = 0; v < 8; ++v) {
            int r = v + half * 8;
            float p = __expf(s[v] - m_i[r]);
            s[v] = p;
            Pbuf[r][wave * 16 + col0] = (bf16_t)p;
            float su = p;
            su += __shfl_xor(su, 1, 32);
            su += __shfl_xor(su, 2, 32);
            su += __shfl_xor(su, 4, 32);
            su += __shfl_xor(su, 8, 32);
            if (col0 == 0) red[wave][r] = su;
        }
        __syncthreads();

        if (threadIdx.x < 16) {
            int r = threadIdx.x;
            l_i[r] = l_i[r] * corr[r] + red[0][r] + red[1][r] + red[2][r] + red[3][r];
        }

        // -------- rescale accumulators and do PV (K = 64), ping-pong --------
        float cr[8];
        #pragma unroll
        for (int v = 0; v < 8; ++v) cr[v] = corr[v + half * 8];
        #pragma unroll
        for (int t = 0; t < 8; ++t)
            #pragma unroll
            for (int v = 0; v < 8; ++v) acc[t][v] *= cr[v];

        #pragma unroll
        for (int kk = 0; kk < 2; ++kk) {
            // A-layout read of P from LDS: lane row = col0, K split by half
            const bf16_t* prow = &Pbuf[col0][kk * 32 + half * 8];
            v8bf plo = *(const v8bf*)prow;
            v8bf phi = *(const v8bf*)(prow + 16);
            v16bf ap = __builtin_shufflevector(plo, phi, 0,1,2,3,4,5,6,7,8,9,10,11,12,13,14,15);

            const bf16_t* Bbase = histT + (size_t)(wave * 128) * N_HIST + j * 64 + kk * 32;
            v16bf bt0 = load_nt(Bbase, N_HIST, lane);
            v16bf bt1;
            #pragma unroll
            for (int t = 0; t < 6; t += 2) {
                bt1 = load_nt(Bbase + (size_t)(t + 1) * 16 * N_HIST, N_HIST, lane);
                acc[t] = wmma_bf16(ap, bt0, acc[t]);
                bt0 = load_nt(Bbase + (size_t)(t + 2) * 16 * N_HIST, N_HIST, lane);
                acc[t + 1] = wmma_bf16(ap, bt1, acc[t + 1]);
            }
            bt1 = load_nt(Bbase + (size_t)7 * 16 * N_HIST, N_HIST, lane);
            acc[6] = wmma_bf16(ap, bt0, acc[6]);
            acc[7] = wmma_bf16(ap, bt1, acc[7]);
        }
        __syncthreads();
    }

    // -------- finalize: out = acc / l_i, bf16 --------
    float li[8];
    #pragma unroll
    for (int v = 0; v < 8; ++v) li[v] = l_i[v + half * 8];
    #pragma unroll
    for (int t = 0; t < 8; ++t) {
        int d = wave * 128 + t * 16 + col0;
        #pragma unroll
        for (int v = 0; v < 8; ++v) {
            int r = m0 + v + half * 8;
            lvlout[((size_t)l * N_CUR + r) * D_FEAT + d] = (bf16_t)(acc[t][v] / li[v]);
        }
    }
}

// ---------------------------------------------------------------------------
// Final GEMM: out[4096,512](f32) = concat(lvlout)[4096,1536] * Wf^T + bf.
// K = 3 segments of 512 (level-major concat), flattened into one 48-step
// ping-pong pipelined loop. Each wave computes 32(M) x 64(N).
// ---------------------------------------------------------------------------
__global__ __launch_bounds__(128) void gemm_final(
    const bf16_t* __restrict__ lvl, const bf16_t* __restrict__ Wf,
    const float* __restrict__ bias, float* __restrict__ out)
{
    int lane = threadIdx.x & 31;
    int wave = threadIdx.x >> 5;
    int m0 = (blockIdx.y * 4 + wave) * 32;
    int n0 = blockIdx.x * 64;

    v8f acc[2][4];
    #pragma unroll
    for (int u = 0; u < 2; ++u)
        #pragma unroll
        for (int t = 0; t < 4; ++t)
            #pragma unroll
            for (int j = 0; j < 8; ++j) acc[u][t][j] = 0.f;

    auto loadF = [&](v16bf& a0, v16bf& a1, v16bf* b, int ks) {
        int lv = ks >> 4, k = (ks & 15) * 32;
        a0 = load_nt(lvl + ((size_t)lv * N_CUR + m0) * D_FEAT + k, D_FEAT, lane);
        a1 = load_nt(lvl + ((size_t)lv * N_CUR + m0 + 16) * D_FEAT + k, D_FEAT, lane);
        #pragma unroll
        for (int t = 0; t < 4; ++t)
            b[t] = load_nt(Wf + (size_t)(n0 + t * 16) * (L_LVL * D_FEAT) + lv * D_FEAT + k,
                           L_LVL * D_FEAT, lane);
    };
    auto mmaF = [&](v16bf& a0, v16bf& a1, v16bf* b) {
        #pragma unroll
        for (int t = 0; t < 4; ++t) {
            acc[0][t] = wmma_bf16(a0, b[t], acc[0][t]);
            acc[1][t] = wmma_bf16(a1, b[t], acc[1][t]);
        }
    };

    v16bf a0x, a1x, bx[4], a0y, a1y, by[4];
    loadF(a0x, a1x, bx, 0);
    int ks = 0;
    for (; ks < 46; ks += 2) {
        loadF(a0y, a1y, by, ks + 1);
        mmaF(a0x, a1x, bx);
        loadF(a0x, a1x, bx, ks + 2);
        mmaF(a0y, a1y, by);
    }
    loadF(a0y, a1y, by, 47);
    mmaF(a0x, a1x, bx);
    mmaF(a0y, a1y, by);

    int col0 = lane & 15, half = (lane >> 4) & 1;
    #pragma unroll
    for (int u = 0; u < 2; ++u)
        #pragma unroll
        for (int t = 0; t < 4; ++t) {
            int col = n0 + t * 16 + col0;
            float bv = bias[col];
            #pragma unroll
            for (int v = 0; v < 8; ++v) {
                int r = m0 + u * 16 + v + half * 8;
                out[(size_t)r * D_FEAT + col] = acc[u][t][v] + bv;
            }
        }
}

// ---------------------------------------------------------------------------
extern "C" void kernel_launch(void* const* d_in, const int* in_sizes, int n_in,
                              void* d_out, int out_size, void* d_ws, size_t ws_size,
                              hipStream_t stream) {
    (void)in_sizes; (void)n_in; (void)out_size; (void)ws_size;
    const float* xc  = (const float*)d_in[0];
    const float* xh  = (const float*)d_in[1];
    const float* W1  = (const float*)d_in[2];
    const float* b1  = (const float*)d_in[3];
    const float* W2  = (const float*)d_in[4];
    const float* b2  = (const float*)d_in[5];
    const float* Wf  = (const float*)d_in[6];
    const float* bfv = (const float*)d_in[7];
    float* out = (float*)d_out;

    char* ws = (char*)d_ws;
    size_t off = 0;
    auto alloc = [&](size_t bytes) { size_t o = off; off += (bytes + 255) & ~(size_t)255; return (void*)(ws + o); };

    bf16_t* xbf    = (bf16_t*)alloc((size_t)N_ALL * D_FEAT * 2);            // [12288,512]
    bf16_t* histT  = (bf16_t*)alloc((size_t)D_FEAT * N_HIST * 2);           // [512,8192]
    bf16_t* W1bf   = (bf16_t*)alloc((size_t)L_LVL * R_DIM * D_FEAT * 2);
    bf16_t* W2bf   = (bf16_t*)alloc((size_t)L_LVL * D_FEAT * R_DIM * 2);
    bf16_t* Wfbf   = (bf16_t*)alloc((size_t)D_FEAT * L_LVL * D_FEAT * 2);
    bf16_t* hbuf   = (bf16_t*)alloc((size_t)L_LVL * N_ALL * R_DIM * 2);     // [3,12288,128]
    bf16_t* projb  = (bf16_t*)alloc((size_t)L_LVL * N_ALL * D_FEAT * 2);    // [3,12288,512]
    bf16_t* lvlout = (bf16_t*)alloc((size_t)L_LVL * N_CUR * D_FEAT * 2);    // [3,4096,512]

    // 1) converts + transpose
    cvt_f32_bf16<<<2048, 256, 0, stream>>>(xc, xbf, (long)N_CUR * D_FEAT);
    cvt_f32_bf16<<<2048, 256, 0, stream>>>(xh, xbf + (size_t)N_CUR * D_FEAT, (long)N_HIST * D_FEAT);
    cvt_f32_bf16<<<512, 256, 0, stream>>>(W1, W1bf, (long)L_LVL * R_DIM * D_FEAT);
    cvt_f32_bf16<<<512, 256, 0, stream>>>(W2, W2bf, (long)L_LVL * D_FEAT * R_DIM);
    cvt_f32_bf16<<<1024, 256, 0, stream>>>(Wf, Wfbf, (long)D_FEAT * L_LVL * D_FEAT);
    transpose_hist<<<4096, 256, 0, stream>>>(xh, histT);

    // 2) h = relu(x W1^T + b1)   [3,12288,128]
    gemm_nt<<<dim3(R_DIM / 64, N_ALL / 128, L_LVL), 128, 0, stream>>>(
        xbf, W1bf, b1, hbuf,
        /*N=*/R_DIM, /*K=*/D_FEAT, /*lda=*/D_FEAT, /*ldb=*/D_FEAT,
        /*sA=*/0, /*sB=*/(long)R_DIM * D_FEAT, /*sBias=*/R_DIM,
        /*sC=*/(long)N_ALL * R_DIM, /*relu=*/1);

    // 3) proj = h W2^T + b2      [3,12288,512]
    gemm_nt<<<dim3(D_FEAT / 64, N_ALL / 128, L_LVL), 128, 0, stream>>>(
        hbuf, W2bf, b2, projb,
        /*N=*/D_FEAT, /*K=*/R_DIM, /*lda=*/R_DIM, /*ldb=*/R_DIM,
        /*sA=*/(long)N_ALL * R_DIM, /*sB=*/(long)D_FEAT * R_DIM, /*sBias=*/D_FEAT,
        /*sC=*/(long)N_ALL * D_FEAT, /*relu=*/0);

    // 4) flash attention per level -> lvlout [3,4096,512] bf16
    flash_attn<<<dim3(N_CUR / 16, 1, L_LVL), 128, 0, stream>>>(projb, histT, lvlout);

    // 5) out = concat(lvlout) Wf^T + bf   [4096,512] f32
    gemm_final<<<dim3(D_FEAT / 64, N_CUR / 128, 1), 128, 0, stream>>>(lvlout, Wfbf, bfv, out);
}
